// SLFusedProjLoss_90503550861754
// MI455X (gfx1250) — matrix operand order repaired and checked by reading, not compiled, not run on hardware
//
#include <hip/hip_runtime.h>
#include <hip/hip_bf16.h>
#include <stdint.h>

#define IGNORE_INDEX (-100)

typedef __bf16 bf16_t;
typedef __attribute__((ext_vector_type(16))) __bf16 v16bf;
typedef __attribute__((ext_vector_type(8)))  __bf16 v8bf;
typedef __attribute__((ext_vector_type(4)))  __bf16 v4bf;
typedef __attribute__((ext_vector_type(8)))  float  v8f;

static constexpr int kN   = 8192;
static constexpr int kD   = 2048;
static constexpr int kV   = 32000;
static constexpr int ROWS = 32;            // tokens per workgroup (2 x 16-row WMMA tiles)
static constexpr int COLS = 128;           // vocab tile per outer step (8 waves x 16 cols)
static constexpr int BK   = 32;            // bf16 WMMA K
static constexpr int NB   = kN / ROWS;     // 256 workgroups
static constexpr int NCW  = 8;             // column-waves per block

// LDS x-tile swizzle: rows r and r+16 interleaved so both row-tiles sit within
// one 16-bit DS offset window of a single base pointer.
//   phys(row, kByte) = (row & 15) * 8192 + (row >> 4) * 4096 + kByte
__device__ __forceinline__ int xswz(int row) {
  return (row & 15) * 8192 + (row >> 4) * 4096;
}

union AB { v16bf v; v8bf h[2]; };

// ---------------------------------------------------------------------------
// Kernel 1: W fp32 -> bf16 (128MB bf16 weight image fits the 192MB L2)
// ---------------------------------------------------------------------------
__global__ __launch_bounds__(256) void convert_w_bf16(const float* __restrict__ w,
                                                      bf16_t* __restrict__ wb) {
  const size_t n4 = (size_t)kV * kD / 4;
  const size_t stride = (size_t)gridDim.x * blockDim.x;
  for (size_t i = (size_t)blockIdx.x * blockDim.x + threadIdx.x; i < n4; i += stride) {
    float4 f = ((const float4*)w)[i];
    v4bf b;
    b[0] = (bf16_t)f.x; b[1] = (bf16_t)f.y; b[2] = (bf16_t)f.z; b[3] = (bf16_t)f.w;
    ((v4bf*)wb)[i] = b;
  }
}

// single-exp online-logsumexp merge of running (m,s) with one new logit l
__device__ __forceinline__ void lse_push(float& m, float& s, float l) {
  float nm = fmaxf(m, l);
  float a  = __expf(fminf(m, l) - nm);   // the other factor is exp(0)=1
  s = (l > m) ? (s * a + 1.0f) : (s + a);
  m = nm;
}

// single-exp merge of two running pairs
__device__ __forceinline__ void lse_merge(float& m, float& s, float om, float os) {
  float nm = fmaxf(m, om);
  float a  = __expf(fminf(m, om) - nm);
  s = (om > m) ? (s * a + os) : (s + os * a);
  m = nm;
}

// ---------------------------------------------------------------------------
// Kernel 2: fused projection + online-softmax cross-entropy
// 8 waves each own 16 vocab columns x all 32 rows: each B tile feeds 2 WMMAs.
// ---------------------------------------------------------------------------
template <bool PRECONV>
__global__ __launch_bounds__(256, 1)
void fused_proj_ce(const float* __restrict__ x,
                   const float* __restrict__ w32,
                   const bf16_t* __restrict__ w16,
                   const int* __restrict__ target,
                   float* __restrict__ psum,
                   int* __restrict__ pcnt) {
  extern __shared__ char smem[];
  float*  statM  = (float*)(smem + ROWS * kD * 2);    // [ROWS][NCW]
  float*  statS  = statM + ROWS * NCW;                // [ROWS][NCW]
  float*  tgtLog = statS + ROWS * NCW;                // [ROWS]
  int*    tgts   = (int*)(tgtLog + ROWS);             // [ROWS]

  const int tid      = threadIdx.x;
  const int lane     = tid & 31;
  const int wave     = tid >> 5;         // = column tile 0..7
  const int halfU    = lane >> 4;        // 0 | 1 (upper lane half)
  const int cLane    = lane & 15;
  const int blockRow = blockIdx.x * ROWS;

  // ---- stage x block into LDS as bf16, row-tile-interleaved swizzle ----
  {
    const float4* xg = (const float4*)(x + (size_t)blockRow * kD);
    for (int i = tid; i < ROWS * kD / 4; i += 256) {
      const int row = i >> 9;           // i / (kD/4)
      const int kv  = i & 511;          // i % (kD/4), units of 4 elements
      float4 f = xg[i];
      v4bf b;
      b[0] = (bf16_t)f.x; b[1] = (bf16_t)f.y; b[2] = (bf16_t)f.z; b[3] = (bf16_t)f.w;
      *(v4bf*)(smem + xswz(row) + kv * 8) = b;
    }
  }
  if (tid < ROWS) tgts[tid] = target[blockRow + tid];
  __syncthreads();

  // A-matrix LDS base (contiguous layout: lanes 0-15 take K {0..15};
  // lanes 16-31 take K {16..31} of each 32-K step).
  // Row-tile 0 at +0, row-tile 1 at +4096 (static DS offset).
  const char* aBase = smem + xswz(cLane) + halfU * 32;

  // slot g of acc0 <-> local row halfU*8+g; acc1 <-> 16+halfU*8+g
  const int rBase0 = halfU * 8;
  const int rBase1 = 16 + halfU * 8;

  float m[16], s[16];
#pragma unroll
  for (int g = 0; g < 16; ++g) { m[g] = -3.0e38f; s[g] = 0.0f; }

#pragma unroll 1
  for (int v0 = 0; v0 < kV; v0 += COLS) {
    const int colG = v0 + wave * 16 + cLane;      // vocab row this lane loads (B column)
    v8f acc0 = {}, acc1 = {};                     // row-tile 0 / row-tile 1 chains

    // B-matrix base: lane = column, contiguous 16 K-elements per lane half
    const char* bBase;
    if (PRECONV) bBase = (const char*)(w16 + (size_t)colG * kD + halfU * 16);
    else         bBase = (const char*)(w32 + (size_t)colG * kD + halfU * 16);

    // prefetch next vocab tile's weight row (global_prefetch_b8)
    if (v0 + COLS < kV) {
      const void* pf = PRECONV ? (const void*)(w16 + (size_t)(colG + COLS) * kD)
                               : (const void*)(w32 + (size_t)(colG + COLS) * kD);
      __builtin_prefetch(pf, 0, 1);
    }

#pragma unroll 2
    for (int k0 = 0; k0 < kD; k0 += 2 * BK) {
      AB A00, A01, A10, A11, B0, B1;
      // B once, reused by both row tiles
      if (PRECONV) {
        B0.h[0] = *(const v8bf*)(bBase + k0 * 2);           // global_load_b128
        B0.h[1] = *(const v8bf*)(bBase + k0 * 2 + 16);
        B1.h[0] = *(const v8bf*)(bBase + (k0 + BK) * 2);
        B1.h[1] = *(const v8bf*)(bBase + (k0 + BK) * 2 + 16);
      } else {
        const float4* bf0 = (const float4*)(bBase + k0 * 4);
        const float4* bf1 = (const float4*)(bBase + (k0 + BK) * 4);
#pragma unroll
        for (int q = 0; q < 2; ++q) {
          const float4* bp = q ? bf1 : bf0;
          float4 f0 = bp[0], f1 = bp[1], f2 = bp[2], f3 = bp[3];
          v8bf lo, hi;
          lo[0] = (bf16_t)f0.x; lo[1] = (bf16_t)f0.y; lo[2] = (bf16_t)f0.z; lo[3] = (bf16_t)f0.w;
          lo[4] = (bf16_t)f1.x; lo[5] = (bf16_t)f1.y; lo[6] = (bf16_t)f1.z; lo[7] = (bf16_t)f1.w;
          hi[0] = (bf16_t)f2.x; hi[1] = (bf16_t)f2.y; hi[2] = (bf16_t)f2.z; hi[3] = (bf16_t)f2.w;
          hi[4] = (bf16_t)f3.x; hi[5] = (bf16_t)f3.y; hi[6] = (bf16_t)f3.z; hi[7] = (bf16_t)f3.w;
          if (q) { B1.h[0] = lo; B1.h[1] = hi; } else { B0.h[0] = lo; B0.h[1] = hi; }
        }
      }
      // A tiles: all static DS offsets off one base register
      A00.h[0] = *(const v8bf*)(aBase + k0 * 2);                  // ds_load_b128
      A00.h[1] = *(const v8bf*)(aBase + k0 * 2 + 16);
      A01.h[0] = *(const v8bf*)(aBase + k0 * 2 + 4096);
      A01.h[1] = *(const v8bf*)(aBase + k0 * 2 + 4096 + 16);
      A10.h[0] = *(const v8bf*)(aBase + (k0 + BK) * 2);
      A10.h[1] = *(const v8bf*)(aBase + (k0 + BK) * 2 + 16);
      A11.h[0] = *(const v8bf*)(aBase + (k0 + BK) * 2 + 4096);
      A11.h[1] = *(const v8bf*)(aBase + (k0 + BK) * 2 + 4096 + 16);

      acc0 = __builtin_amdgcn_wmma_f32_16x16x32_bf16(false, A00.v, false, B0.v,
                                                     (short)0, acc0, false, false);
      acc1 = __builtin_amdgcn_wmma_f32_16x16x32_bf16(false, A01.v, false, B0.v,
                                                     (short)0, acc1, false, false);
      acc0 = __builtin_amdgcn_wmma_f32_16x16x32_bf16(false, A10.v, false, B1.v,
                                                     (short)0, acc0, false, false);
      acc1 = __builtin_amdgcn_wmma_f32_16x16x32_bf16(false, A11.v, false, B1.v,
                                                     (short)0, acc1, false, false);
    }

    // online logsumexp update (one exp per logit), one column per slot
#pragma unroll
    for (int g = 0; g < 8; ++g) {
      lse_push(m[g],     s[g],     acc0[g]);
      lse_push(m[8 + g], s[8 + g], acc1[g]);
    }
  }

  // ---- reduce (m,s) across the 16 column-lanes of each half-wave ----
#pragma unroll
  for (int g = 0; g < 16; ++g) {
#pragma unroll
    for (int mask = 1; mask < 16; mask <<= 1) {
      float om = __shfl_xor(m[g], mask, 32);
      float os = __shfl_xor(s[g], mask, 32);
      lse_merge(m[g], s[g], om, os);
    }
  }
  if (cLane == 0) {
#pragma unroll
    for (int g = 0; g < 8; ++g) {
      statM[(rBase0 + g) * NCW + wave] = m[g];
      statS[(rBase0 + g) * NCW + wave] = s[g];
      statM[(rBase1 + g) * NCW + wave] = m[8 + g];
      statS[(rBase1 + g) * NCW + wave] = s[8 + g];
    }
  }

  // ---- epilogue: target logits as direct fp32 dot products (4 rows/wave) ----
  {
#pragma unroll 1
    for (int r = wave * 4; r < wave * 4 + 4; ++r) {
      const int tg     = tgts[r];
      const int safeTg = (tg == IGNORE_INDEX) ? 0 : tg;
      const float4* xr = (const float4*)(x + (size_t)(blockRow + r) * kD);
      const float4* wr = (const float4*)(w32 + (size_t)safeTg * kD);
      float p = 0.0f;
#pragma unroll 4
      for (int i = lane; i < kD / 4; i += 32) {
        float4 xv = xr[i];
        float4 wv = wr[i];
        p += xv.x * wv.x + xv.y * wv.y + xv.z * wv.z + xv.w * wv.w;
      }
#pragma unroll
      for (int mask = 1; mask < 32; mask <<= 1) p += __shfl_xor(p, mask, 32);
      if (lane == 0) tgtLog[r] = p;
    }
  }
  __syncthreads();

  // ---- combine the 8 column-wave partials, compute per-row loss ----
  if (tid < ROWS) {  // exactly wave 0: full-wave shuffles are safe
    float mm = statM[tid * NCW + 0];
    float ss = statS[tid * NCW + 0];
#pragma unroll
    for (int c = 1; c < NCW; ++c) lse_merge(mm, ss, statM[tid * NCW + c], statS[tid * NCW + c]);
    const int  tg    = tgts[tid];
    const bool valid = (tg != IGNORE_INDEX);
    float loss = valid ? (mm + __logf(ss) - tgtLog[tid]) : 0.0f;
    int   cnt  = valid ? 1 : 0;
#pragma unroll
    for (int mask = 1; mask < 32; mask <<= 1) {
      loss += __shfl_xor(loss, mask, 32);
      cnt  += __shfl_xor(cnt, mask, 32);
    }
    if (tid == 0) { psum[blockIdx.x] = loss; pcnt[blockIdx.x] = cnt; }
  }
}

// ---------------------------------------------------------------------------
// Kernel 3: deterministic fixed-order final reduction
// ---------------------------------------------------------------------------
__global__ void finalize_loss(const float* __restrict__ psum,
                              const int* __restrict__ pcnt,
                              float* __restrict__ out) {
  if (blockIdx.x == 0 && threadIdx.x == 0) {
    float s = 0.0f;
    int   c = 0;
    for (int i = 0; i < NB; ++i) { s += psum[i]; c += pcnt[i]; }
    out[0] = s / (float)(c > 0 ? c : 1);
  }
}

// ---------------------------------------------------------------------------
extern "C" void kernel_launch(void* const* d_in, const int* in_sizes, int n_in,
                              void* d_out, int out_size, void* d_ws, size_t ws_size,
                              hipStream_t stream) {
  const float* x   = (const float*)d_in[0];
  const float* w   = (const float*)d_in[1];
  const int*   tgt = (const int*)d_in[2];
  // d_in[3] = num_splits: implementation detail of the reference, ignored.
  float* out = (float*)d_out;

  char*   ws   = (char*)d_ws;
  float*  psum = (float*)ws;                 // NB floats
  int*    pcnt = (int*)(ws + 4096);          // NB ints
  bf16_t* wb   = (bf16_t*)(ws + 8192);       // bf16 weights (128 MB)

  const size_t need  = 8192 + (size_t)kV * kD * sizeof(bf16_t);
  const bool preconv = (ws_size >= need);
  const size_t shmem = (size_t)ROWS * kD * 2       // x tile (bf16, swizzled)
                     + (size_t)ROWS * NCW * 4 * 2  // statM + statS
                     + (size_t)ROWS * 4            // tgtLog
                     + (size_t)ROWS * 4;           // tgts

  if (preconv) {
    convert_w_bf16<<<2048, 256, 0, stream>>>(w, wb);
    (void)hipFuncSetAttribute(reinterpret_cast<const void*>(&fused_proj_ce<true>),
                              hipFuncAttributeMaxDynamicSharedMemorySize, (int)shmem);
    fused_proj_ce<true><<<NB, 256, shmem, stream>>>(x, w, wb, tgt, psum, pcnt);
  } else {
    (void)hipFuncSetAttribute(reinterpret_cast<const void*>(&fused_proj_ce<false>),
                              hipFuncAttributeMaxDynamicSharedMemorySize, (int)shmem);
    fused_proj_ce<false><<<NB, 256, shmem, stream>>>(x, w, wb, tgt, psum, pcnt);
  }
  finalize_loss<<<1, 1, 0, stream>>>(psum, pcnt, out);
}
